// GIN_layer_75531294867868
// MI455X (gfx1250) — matrix operand order repaired and checked by reading, not compiled, not run on hardware
//
#include <hip/hip_runtime.h>
#include <hip/hip_bf16.h>

typedef __attribute__((ext_vector_type(8)))  float  v8f;
typedef __attribute__((ext_vector_type(16))) __bf16 v16bf;
typedef __attribute__((ext_vector_type(8)))  __bf16 v8bf;
typedef int v4i __attribute__((vector_size(16)));

#define BN_EPS 1e-5f

#if __has_builtin(__builtin_amdgcn_global_load_async_to_lds_b128) && \
    __has_builtin(__builtin_amdgcn_s_wait_asynccnt)
#define HAVE_ASYNC 1
#else
#define HAVE_ASYNC 0
#endif

#define GAS __attribute__((address_space(1)))
#define LAS __attribute__((address_space(3)))

__device__ __forceinline__ __bf16 f2bf(float f) {
  unsigned u = __float_as_uint(f);
  u += 0x7FFFu + ((u >> 16) & 1u);            // round-to-nearest-even
  unsigned short h = (unsigned short)(u >> 16);
  return __builtin_bit_cast(__bf16, h);
}
__device__ __forceinline__ float bf2f(__bf16 b) {
  unsigned short h = __builtin_bit_cast(unsigned short, b);
  return __uint_as_float(((unsigned)h) << 16);
}

// ---------------- kernel 0: h = (1+eps)*x ; zero stats ----------------
__global__ __launch_bounds__(256) void k_init(const float* __restrict__ x,
                                              const float* __restrict__ eps,
                                              float* __restrict__ h,
                                              float* __restrict__ stats,
                                              long total4) {
  long i = (long)blockIdx.x * 256 + threadIdx.x;
  float e = 1.0f + eps[0];
  if (i < total4) {
    float4 v = ((const float4*)x)[i];
    v.x *= e; v.y *= e; v.z *= e; v.w *= e;
    ((float4*)h)[i] = v;
  }
  if (i < 1024) stats[i] = 0.0f;  // sum1/sumsq1/sum2/sumsq2
}

// ---------------- kernel 1: edge scatter-add (1 wave per edge) ----------------
__global__ __launch_bounds__(256) void k_scatter(const float* __restrict__ x,
                                                 const int* __restrict__ ei,
                                                 float* __restrict__ h, int E) {
  int wave = (int)(((unsigned)blockIdx.x * 256u + threadIdx.x) >> 5);
  int lane = threadIdx.x & 31;
  if (wave >= E) return;
  int src = ei[wave];
  int dst = ei[E + wave];
  float4 v = *(const float4*)(x + (size_t)src * 128 + lane * 4);
  float* hp = h + (size_t)dst * 128 + lane * 4;
  atomicAdd(hp + 0, v.x);
  atomicAdd(hp + 1, v.y);
  atomicAdd(hp + 2, v.z);
  atomicAdd(hp + 3, v.w);
}

// ---------------- f32 -> bf16 bulk convert (8 elems / thread) ----------------
__global__ __launch_bounds__(256) void k_tobf(const float* __restrict__ in,
                                              __bf16* __restrict__ out, long total8) {
  long i = (long)blockIdx.x * 256 + threadIdx.x;
  if (i >= total8) return;
  const float4* p = (const float4*)(in + i * 8);
  float4 a = p[0], b = p[1];
  v8bf v;
  v[0] = f2bf(a.x); v[1] = f2bf(a.y); v[2] = f2bf(a.z); v[3] = f2bf(a.w);
  v[4] = f2bf(b.x); v[5] = f2bf(b.y); v[6] = f2bf(b.z); v[7] = f2bf(b.w);
  *(v8bf*)(out + i * 8) = v;
}

// ------- weight convert+transpose: W[k][c] f32 -> Wc[c*K + k] bf16 -------
__global__ __launch_bounds__(256) void k_wconv(const float* __restrict__ W,
                                               __bf16* __restrict__ Wc, int K) {
  int idx = blockIdx.x * 256 + threadIdx.x;
  if (idx >= K * 256) return;
  int k = idx >> 8;
  int c = idx & 255;
  Wc[(size_t)c * K + k] = f2bf(W[idx]);
}

// ---- in-place BN1 + ReLU on bf16 activations: a = bf16(relu(a*sc+sh)) ----
__global__ __launch_bounds__(256) void k_a1norm(__bf16* __restrict__ a,
                                                const float* __restrict__ scale,
                                                const float* __restrict__ shift,
                                                long total8) {
  long i = (long)blockIdx.x * 256 + threadIdx.x;
  if (i >= total8) return;
  __bf16* p = a + i * 8;
  v8bf v = *(v8bf*)p;
  int c = (int)((i * 8) & 255);
#pragma unroll
  for (int j = 0; j < 8; ++j) {
    float f = bf2f(v[j]) * scale[c + j] + shift[c + j];
    v[j] = f2bf(fmaxf(f, 0.0f));
  }
  *(v8bf*)p = v;
}

// ---------------- tile staging: pure bf16 copy, async when available -------
static constexpr int LSTR = 40;  // padded LDS stride (bf16 elems), conflict-free

__device__ __forceinline__ void stage_tile(const __bf16* __restrict__ Abf,
                                           const __bf16* __restrict__ Wc,
                                           __bf16* __restrict__ a_lds,
                                           __bf16* __restrict__ b_lds,
                                           int tid, int rowBase, int N, int K,
                                           int kt) {
  // A tile: 128 rows x 32 bf16 (64B per row -> 4 x 16B chunks)
#pragma unroll
  for (int i = 0; i < 2; ++i) {
    int idx = tid + i * 256;          // 0..511
    int row = idx >> 2;
    int c8 = (idx & 3) << 3;
    int g_row = rowBase + row;
    __bf16* ldst = a_lds + row * LSTR + c8;
    if (g_row < N) {
      const __bf16* gsrc = Abf + (size_t)g_row * K + kt * 32 + c8;
#if HAVE_ASYNC
      __builtin_amdgcn_global_load_async_to_lds_b128((GAS v4i*)gsrc,
                                                     (LAS v4i*)ldst, 0, 0);
#else
      *(uint4*)ldst = *(const uint4*)gsrc;
#endif
    } else {
      uint4 z = make_uint4(0u, 0u, 0u, 0u);
      *(uint4*)ldst = z;              // zero pad rows -> zero stats contribution
    }
  }
  // B tile: 256 cols x 32 bf16, Wc is already column-major bf16
#pragma unroll
  for (int i = 0; i < 4; ++i) {
    int idx = tid + i * 256;          // 0..1023
    int col = idx >> 2;
    int p8 = (idx & 3) << 3;
    const __bf16* gsrc = Wc + (size_t)col * K + kt * 32 + p8;
    __bf16* ldst = b_lds + col * LSTR + p8;
#if HAVE_ASYNC
    __builtin_amdgcn_global_load_async_to_lds_b128((GAS v4i*)gsrc,
                                                   (LAS v4i*)ldst, 0, 0);
#else
    *(uint4*)ldst = *(const uint4*)gsrc;
#endif
  }
}

// ---------------- GEMM + fused BN-stats ----------------
// A bf16 [N x K], Wc bf16 column-major [256 x K].
// MODE 1: out = bf16 raw activations.  MODE 2: out = f32 raw (into d_out).
template <int MODE>
__global__ __launch_bounds__(256) void k_gemm(const __bf16* __restrict__ Abf,
                                              const __bf16* __restrict__ Wc,
                                              __bf16* __restrict__ outB,
                                              float* __restrict__ outF,
                                              float* __restrict__ stats,
                                              int N, int K) {
  __shared__ __bf16 a_lds[2][128 * LSTR];   // 2 x 10240 B
  __shared__ __bf16 b_lds[2][256 * LSTR];   // 2 x 20480 B
  __shared__ float s_lds[256];
  __shared__ float ss_lds[256];

  const int tid = threadIdx.x;
  const int lane = tid & 31;
  const int wv = tid >> 5;
  const int rowBase = blockIdx.x * 128;
  const bool hiHalf = lane >= 16;
  const int n16 = lane & 15;

  s_lds[tid] = 0.0f;
  ss_lds[tid] = 0.0f;

  v8f acc[16];
#pragma unroll
  for (int i = 0; i < 16; ++i) acc[i] = (v8f)0.0f;

  const int ktiles = K >> 5;
  stage_tile(Abf, Wc, a_lds[0], b_lds[0], tid, rowBase, N, K, 0);

  for (int kt = 0; kt < ktiles; ++kt) {
    const int b = kt & 1;
#if HAVE_ASYNC
    __builtin_amdgcn_s_wait_asynccnt(0);
#endif
    __syncthreads();
    if (kt + 1 < ktiles)  // prefetch next tile; overlaps with WMMA below
      stage_tile(Abf, Wc, a_lds[(kt + 1) & 1], b_lds[(kt + 1) & 1], tid,
                 rowBase, N, K, kt + 1);

    // ---- WMMA: each wave owns 16 rows x 256 cols ----
    const __bf16* ap = &a_lds[b][(wv * 16 + n16) * LSTR + (hiHalf ? 8 : 0)];
    v8bf alo = *(const v8bf*)(ap);
    v8bf ahi = *(const v8bf*)(ap + 16);
    v16bf afrag = __builtin_shufflevector(alo, ahi, 0, 1, 2, 3, 4, 5, 6, 7,
                                          8, 9, 10, 11, 12, 13, 14, 15);
    const int kb = hiHalf ? 8 : 0;
#pragma unroll
    for (int nt = 0; nt < 16; ++nt) {
      const __bf16* bp = &b_lds[b][(nt * 16 + n16) * LSTR + kb];
      v8bf blo = *(const v8bf*)(bp);
      v8bf bhi = *(const v8bf*)(bp + 16);
      v16bf bfrag = __builtin_shufflevector(blo, bhi, 0, 1, 2, 3, 4, 5, 6, 7,
                                            8, 9, 10, 11, 12, 13, 14, 15);
      acc[nt] = __builtin_amdgcn_wmma_f32_16x16x32_bf16(
          false, afrag, false, bfrag, (short)0, acc[nt], false, false);
    }
  }

  // ---- epilogue: store raw output + fused column stats ----
  const int mbase = rowBase + wv * 16 + (hiHalf ? 8 : 0);
#pragma unroll
  for (int nt = 0; nt < 16; ++nt) {
    int col = nt * 16 + n16;
    float s = 0.f, ss = 0.f;
#pragma unroll
    for (int v = 0; v < 8; ++v) {
      float val = acc[nt][v];
      s += val;
      ss += val * val;
      int g_row = mbase + v;
      if (g_row < N) {
        if (MODE == 1) outB[(size_t)g_row * 256 + col] = f2bf(val);
        else           outF[(size_t)g_row * 256 + col] = val;
      }
    }
    atomicAdd(&s_lds[col], s);
    atomicAdd(&ss_lds[col], ss);
  }
  __syncthreads();
  atomicAdd(&stats[tid], s_lds[tid]);
  atomicAdd(&stats[256 + tid], ss_lds[tid]);
}

// ---------------- BN finalize: stats -> (scale, shift) ----------------
__global__ __launch_bounds__(256) void k_bnfin(const float* __restrict__ stats,
                                               const float* __restrict__ gamma,
                                               const float* __restrict__ beta,
                                               float* __restrict__ scale,
                                               float* __restrict__ shift,
                                               float invN) {
  int t = threadIdx.x;
  float mean = stats[t] * invN;
  float var = stats[256 + t] * invN - mean * mean;
  float inv = rsqrtf(var + BN_EPS);
  float sc = gamma[t] * inv;
  scale[t] = sc;
  shift[t] = beta[t] - mean * sc;
}

// ---------------- in-place BN2 + ReLU on d_out ----------------
__global__ __launch_bounds__(256) void k_outnorm(float* __restrict__ out,
                                                 const float* __restrict__ scale,
                                                 const float* __restrict__ shift,
                                                 long total4) {
  long i = (long)blockIdx.x * 256 + threadIdx.x;
  if (i >= total4) return;
  float4 v = ((const float4*)out)[i];
  int c = (int)((i * 4) & 255);
  v.x = fmaxf(v.x * scale[c + 0] + shift[c + 0], 0.f);
  v.y = fmaxf(v.y * scale[c + 1] + shift[c + 1], 0.f);
  v.z = fmaxf(v.z * scale[c + 2] + shift[c + 2], 0.f);
  v.w = fmaxf(v.w * scale[c + 3] + shift[c + 3], 0.f);
  ((float4*)out)[i] = v;
}

extern "C" void kernel_launch(void* const* d_in, const int* in_sizes, int n_in,
                              void* d_out, int out_size, void* d_ws, size_t ws_size,
                              hipStream_t stream) {
  const float* x   = (const float*)d_in[0];
  const int*   ei  = (const int*)d_in[1];
  const float* eps = (const float*)d_in[3];
  const float* W1  = (const float*)d_in[4];
  const float* g1  = (const float*)d_in[5];
  const float* b1  = (const float*)d_in[6];
  const float* W2  = (const float*)d_in[7];
  const float* g2  = (const float*)d_in[8];
  const float* b2  = (const float*)d_in[9];
  float* out = (float*)d_out;

  const int D = 128, H = 256;
  const int N = in_sizes[0] / D;
  const int E = in_sizes[1] / 2;

  char* ws = (char*)d_ws;
  size_t off = 0;
  float*  h32 = (float*)(ws + off);  off += (size_t)N * D * sizeof(float);
  __bf16* hbf = (__bf16*)(ws + off); off += (size_t)N * D * sizeof(__bf16);
  __bf16* a1  = (__bf16*)(ws + off); off += (size_t)N * H * sizeof(__bf16);
  __bf16* wc1 = (__bf16*)(ws + off); off += (size_t)D * H * sizeof(__bf16);
  __bf16* wc2 = (__bf16*)(ws + off); off += (size_t)H * H * sizeof(__bf16);
  float*  meta = (float*)(ws + off);
  float* stats1 = meta;           // [512] (sum, sumsq)
  float* stats2 = meta + 512;     // [512]
  float* scale1 = meta + 1024;
  float* shift1 = meta + 1280;
  float* scale2 = meta + 1536;
  float* shift2 = meta + 1792;

  long total4h = (long)N * D / 4;
  k_init<<<(int)((total4h + 255) / 256), 256, 0, stream>>>(x, eps, h32, meta, total4h);

  k_scatter<<<(E + 7) / 8, 256, 0, stream>>>(x, ei, h32, E);

  long total8h = (long)N * D / 8;
  k_tobf<<<(int)((total8h + 255) / 256), 256, 0, stream>>>(h32, hbf, total8h);
  k_wconv<<<(D * H + 255) / 256, 256, 0, stream>>>(W1, wc1, D);
  k_wconv<<<(H * H + 255) / 256, 256, 0, stream>>>(W2, wc2, H);

  int gblocks = (N + 127) / 128;
  k_gemm<1><<<gblocks, 256, 0, stream>>>(hbf, wc1, a1, nullptr, stats1, N, D);
  k_bnfin<<<1, 256, 0, stream>>>(stats1, g1, b1, scale1, shift1, 1.0f / (float)N);

  long total8a = (long)N * H / 8;
  k_a1norm<<<(int)((total8a + 255) / 256), 256, 0, stream>>>(a1, scale1, shift1, total8a);

  k_gemm<2><<<gblocks, 256, 0, stream>>>(a1, wc2, nullptr, out, stats2, N, H);
  k_bnfin<<<1, 256, 0, stream>>>(stats2, g2, b2, scale2, shift2, 1.0f / (float)N);

  long total4o = (long)N * H / 4;
  k_outnorm<<<(int)((total4o + 255) / 256), 256, 0, stream>>>(out, scale2, shift2, total4o);
}
